// MolecularRectifiedFlow_44409961840993
// MI455X (gfx1250) — compile-verified, matrix operand order checked by reading, not run on hardware
//
#include <hip/hip_runtime.h>

// ---------------------------------------------------------------------------
// Problem constants (match reference)
// ---------------------------------------------------------------------------
#define NN    65536
#define EE    524288
#define GG    8192
#define HID   512
#define HALF  256          // HID/2
#define SLICES 4           // j-dimension split for clash kernel

typedef __bf16  bf16_t;
typedef __attribute__((ext_vector_type(16))) __bf16 v16bf;
typedef __attribute__((ext_vector_type(8)))  float  v8f;
typedef __attribute__((ext_vector_type(4)))  int    v4i;

typedef __attribute__((address_space(1))) v4i* gv4i_ptr;   // global
typedef __attribute__((address_space(3))) v4i* lv4i_ptr;   // LDS

#if defined(__has_builtin)
# if __has_builtin(__builtin_amdgcn_global_load_async_to_lds_b128)
#  define HAVE_ASYNC_LDS 1
# endif
# if __has_builtin(__builtin_amdgcn_s_wait_asynccnt)
#  define HAVE_WAIT_ASYNC 1
# endif
#endif

// ---------------------------------------------------------------------------
// Workspace layout (bytes, all 256-aligned)
// ---------------------------------------------------------------------------
#define WS_PACKB      0                         // 16*16*32*8 dwords = 256 KB
#define WS_BONDW      (WS_PACKB   + 262144)     // N floats
#define WS_CLASHW     (WS_BONDW   + 262144)     // N floats
#define WS_MASK       (WS_CLASHW  + 262144)     // N ints
#define WS_BONDCORR   (WS_MASK    + 262144)     // N*3 floats
#define WS_XG         (WS_BONDCORR+ 786432)     // G*3 floats
#define WS_CLASHCORR  (WS_XG      + 98304)      // G*3 floats

// ---------------------------------------------------------------------------
// K0: zero accumulators / mask
// ---------------------------------------------------------------------------
__global__ void k_zero(float* bond_corr, float* clash_corr, int* mask) {
    int t = blockIdx.x * 256 + threadIdx.x;
    if (t < NN * 3) bond_corr[t] = 0.0f;
    if (t < NN)     mask[t]      = 0;
    if (t < GG * 3) clash_corr[t] = 0.0f;
}

// K0b: scatter gen mask + gather xg = x_t[gen_idx]
__global__ void k_scatter(const int* __restrict__ gen_idx,
                          const float* __restrict__ x_t,
                          int* __restrict__ mask, float* __restrict__ xg) {
    int g = blockIdx.x * 256 + threadIdx.x;
    if (g >= GG) return;
    int idx = gen_idx[g];
    mask[idx] = 1;
    xg[g * 3 + 0] = x_t[idx * 3 + 0];
    xg[g * 3 + 1] = x_t[idx * 3 + 1];
    xg[g * 3 + 2] = x_t[idx * 3 + 2];
}

// ---------------------------------------------------------------------------
// K1: pack W_proj (fp32 [512,256]) into bf16 WMMA B-fragments.
// Fragment (kc, nt): 32 lanes x 8 dwords, lane-major so the GEMM kernel does
// two b128 loads per lane. Layout mirrors the A-matrix ISA layout:
//   lane<16 : n = lane,    K pairs {0..7} in v0..3, {16..23} in v4..7
//   lane>=16: n = lane-16, K pairs {8..15} in v0..3, {24..31} in v4..7
// ---------------------------------------------------------------------------
__global__ void k_pack_w(const float* __restrict__ W, bf16_t* __restrict__ packB) {
    int t = blockIdx.x * 256 + threadIdx.x;          // 0 .. 16*16*32-1
    if (t >= 16 * 16 * 32) return;
    int lane = t & 31;
    int nt   = (t >> 5) & 15;
    int kc   = t >> 9;
    int n     = nt * 16 + (lane & 15);
    int kbase = kc * 32 + ((lane >> 4) ? 8 : 0);
    bf16_t* dst = packB + (size_t)t * 16;
    #pragma unroll
    for (int v = 0; v < 8; ++v) {
        int k0 = kbase + (v < 4 ? 2 * v : 16 + 2 * (v - 4));
        dst[2 * v + 0] = (bf16_t)W[(size_t)(k0 + 0) * HALF + n];
        dst[2 * v + 1] = (bf16_t)W[(size_t)(k0 + 1) * HALF + n];
    }
}

// ---------------------------------------------------------------------------
// GEMM helpers
// ---------------------------------------------------------------------------
struct ARaw { float4 a, b, c, d; };   // 16 fp32 for one A fragment

static __device__ __forceinline__ void load_araw(const float* __restrict__ H,
                                                 int r, int kc, int koff, ARaw& o) {
    const float4* q = reinterpret_cast<const float4*>(H + (size_t)r * HID + kc * 32 + koff);
    o.a = q[0]; o.b = q[1];      // K offsets  0..7  (of this lane's half)
    o.c = q[4]; o.d = q[5];      // K offsets 16..23
}

static __device__ __forceinline__ v16bf cvt_afrag(const ARaw& r) {
    v16bf v;
    v[0]  = (bf16_t)r.a.x; v[1]  = (bf16_t)r.a.y; v[2]  = (bf16_t)r.a.z; v[3]  = (bf16_t)r.a.w;
    v[4]  = (bf16_t)r.b.x; v[5]  = (bf16_t)r.b.y; v[6]  = (bf16_t)r.b.z; v[7]  = (bf16_t)r.b.w;
    v[8]  = (bf16_t)r.c.x; v[9]  = (bf16_t)r.c.y; v[10] = (bf16_t)r.c.z; v[11] = (bf16_t)r.c.w;
    v[12] = (bf16_t)r.d.x; v[13] = (bf16_t)r.d.y; v[14] = (bf16_t)r.d.z; v[15] = (bf16_t)r.d.w;
    return v;
}

union BBuf { uint4 u[2]; v16bf f; };

static __device__ __forceinline__ void load_bfrag(const uint4* __restrict__ lbs,
                                                  int kcl, int nt, int lane, BBuf& b) {
    const uint4* p = lbs + ((size_t)((kcl * 16 + nt) * 32 + lane)) * 2;
    b.u[0] = p[0];
    b.u[1] = p[1];
}

// Cooperative fill of one 128 KB B phase into LDS (8192 uint4).
// Uses the gfx1250 async global->LDS datapath when available (ASYNCcnt).
static __device__ __forceinline__ void fill_b_phase(uint4* lds, const uint4* src, int tid) {
#ifdef HAVE_ASYNC_LDS
    #pragma unroll
    for (int it = 0; it < 32; ++it) {
        int idx = it * 256 + tid;
        __builtin_amdgcn_global_load_async_to_lds_b128(
            (gv4i_ptr)(uintptr_t)(src + idx),
            (lv4i_ptr)(unsigned)(uintptr_t)(lds + idx),
            0, 0);
    }
# ifdef HAVE_WAIT_ASYNC
    __builtin_amdgcn_s_wait_asynccnt(0);
# else
    asm volatile("s_wait_asynccnt 0x0" ::: "memory");
# endif
#else
    #pragma unroll 4
    for (int it = 0; it < 32; ++it) {
        int idx = it * 256 + tid;
        lds[idx] = src[idx];
    }
#endif
}

// ---------------------------------------------------------------------------
// K2: fused WMMA GEMM + dual linear head + sigmoid.
// 8 waves / block, each wave owns a 16-row band, 16 accumulator tiles cover
// all 256 output cols, single K pass (bf16 16x16x32, fp32 accumulate).
//  - B staged in LDS in two 128 KB phases (shared by all 8 waves),
//    fragments double-buffered out of LDS (ds_load_b128).
//  - A register-pipelined: raw fp32 for kc+1 issued before the 16 WMMAs of kc
//    so the HBM stream (the real bottleneck, 134 MB) stays latency-hidden.
//  - feat never touches memory: epilogue dots each tile with w_bond / w_val.
// ---------------------------------------------------------------------------
__global__ __launch_bounds__(256) void k_gemm_sigmoid(
    const float* __restrict__ H, const bf16_t* __restrict__ packB,
    const float* __restrict__ b_proj,
    const float* __restrict__ w_bond, const float* __restrict__ b_bond,
    const float* __restrict__ w_val,  const float* __restrict__ b_val,
    float* __restrict__ bond_w, float* __restrict__ clash_w)
{
    extern __shared__ uint4 lbs[];               // 8192 uint4 = 128 KB

    const int tid     = threadIdx.x;
    const int lane    = tid & 31;
    const int wave    = tid >> 5;
    const int tileRow = blockIdx.x * 128 + wave * 16;
    const int r       = tileRow + (lane & 15);
    const int koff    = (lane >> 4) ? 8 : 0;

    v8f zero = {0.f, 0.f, 0.f, 0.f, 0.f, 0.f, 0.f, 0.f};
    v8f acc[16];
    #pragma unroll
    for (int nt = 0; nt < 16; ++nt) acc[nt] = zero;

    const uint4* pbGlobal = reinterpret_cast<const uint4*>(packB);

    ARaw cur, nxt;
    load_araw(H, r, 0, koff, cur);               // prime A pipeline (kc = 0)

    for (int p = 0; p < 2; ++p) {
        __syncthreads();                          // previous phase fully consumed
        fill_b_phase(lbs, pbGlobal + (size_t)p * 8192, tid);
        __syncthreads();                          // B phase visible to all waves

        for (int kcl = 0; kcl < 8; ++kcl) {
            const int kc = p * 8 + kcl;
            // Prefetch A for next kc (clamped; carries across the phase seam).
            const int kpf = (kc + 1 < 16) ? (kc + 1) : 15;
            load_araw(H, r, kpf, koff, nxt);

            const v16bf a = cvt_afrag(cur);

            BBuf b0, b1;
            load_bfrag(lbs, kcl, 0, lane, b0);
            #pragma unroll
            for (int nt = 0; nt < 16; ++nt) {
                BBuf& bc = (nt & 1) ? b1 : b0;
                BBuf& bn = (nt & 1) ? b0 : b1;
                if (nt < 15) load_bfrag(lbs, kcl, nt + 1, lane, bn);
                acc[nt] = __builtin_amdgcn_wmma_f32_16x16x32_bf16(
                    false, a, false, bc.f, (short)0, acc[nt], false, false);
            }
            cur = nxt;
        }
    }

    // Epilogue: per lane the C-tile column is fixed (col = nt*16 + lane&15),
    // rows vary with the accumulator dword. Dot with both heads, then reduce
    // across the 16 lanes of each half-wave.
    const int nlane = lane & 15;
    float bs[8], vs[8];
    #pragma unroll
    for (int v = 0; v < 8; ++v) { bs[v] = 0.f; vs[v] = 0.f; }

    #pragma unroll
    for (int nt = 0; nt < 16; ++nt) {
        int col = nt * 16 + nlane;
        float wb = w_bond[col], wv = w_val[col], bp = b_proj[col];
        #pragma unroll
        for (int v = 0; v < 8; ++v) {
            float t = acc[nt][v] + bp;       // feat element (row, col)
            bs[v] = fmaf(t, wb, bs[v]);
            vs[v] = fmaf(t, wv, vs[v]);
        }
    }

    #pragma unroll
    for (int v = 0; v < 8; ++v) {
        #pragma unroll
        for (int m = 1; m < 16; m <<= 1) {
            bs[v] += __shfl_xor(bs[v], m, 32);
            vs[v] += __shfl_xor(vs[v], m, 32);
        }
    }

    if (nlane == 0) {
        int rbase = tileRow + ((lane >> 4) ? 8 : 0);
        float bb0 = b_bond[0], bv0 = b_val[0];
        #pragma unroll
        for (int v = 0; v < 8; ++v) {
            int row = rbase + v;
            bond_w[row]  = 1.0f / (1.0f + __expf(-(bs[v] + bb0)));
            clash_w[row] = 1.0f / (1.0f + __expf(-(vs[v] + bv0)));
        }
    }
}

// ---------------------------------------------------------------------------
// K3: bond corrections — per-edge force, scatter with float atomics.
// ---------------------------------------------------------------------------
__global__ void k_bond(const long long* __restrict__ edges,
                       const float* __restrict__ x_t,
                       const int* __restrict__ mask,
                       float* __restrict__ bond_corr)
{
    int i = blockIdx.x * 256 + threadIdx.x;
    if (i >= EE) return;
    long long r = edges[i];
    long long c = edges[(size_t)EE + i];
    if (!(mask[r] | mask[c])) return;     // gate==0 -> zero contribution

    float dx = x_t[r * 3 + 0] - x_t[c * 3 + 0];
    float dy = x_t[r * 3 + 1] - x_t[c * 3 + 1];
    float dz = x_t[r * 3 + 2] - x_t[c * 3 + 2];
    float d2   = fmaxf(dx * dx + dy * dy + dz * dz, 1e-24f);
    float dist = sqrtf(d2) + 1e-8f;
    float s    = (dist - 1.5f) / (dist * dist);

    atomicAdd(&bond_corr[r * 3 + 0],  s * dx);
    atomicAdd(&bond_corr[r * 3 + 1],  s * dy);
    atomicAdd(&bond_corr[r * 3 + 2],  s * dz);
    atomicAdd(&bond_corr[c * 3 + 0], -s * dx);
    atomicAdd(&bond_corr[c * 3 + 1], -s * dy);
    atomicAdd(&bond_corr[c * 3 + 2], -s * dz);
}

// ---------------------------------------------------------------------------
// K4: clash corrections — G x G pairwise, i per thread, j sliced across
// blockIdx.y and staged through LDS in 256-wide chunks.
// ---------------------------------------------------------------------------
__global__ __launch_bounds__(256) void k_clash(const float* __restrict__ xg,
                                               float* __restrict__ clash_corr)
{
    __shared__ float sx[256], sy[256], sz[256];
    const int i     = blockIdx.x * 256 + threadIdx.x;
    const int jbase = blockIdx.y * (GG / SLICES);

    float xi = xg[i * 3 + 0];
    float yi = xg[i * 3 + 1];
    float zi = xg[i * 3 + 2];
    float ax = 0.f, ay = 0.f, az = 0.f;

    for (int chunk = 0; chunk < GG / SLICES; chunk += 256) {
        int j = jbase + chunk + threadIdx.x;
        sx[threadIdx.x] = xg[j * 3 + 0];
        sy[threadIdx.x] = xg[j * 3 + 1];
        sz[threadIdx.x] = xg[j * 3 + 2];
        __syncthreads();
        #pragma unroll 8
        for (int jj = 0; jj < 256; ++jj) {
            int jg = jbase + chunk + jj;
            float dx = xi - sx[jj];
            float dy = yi - sy[jj];
            float dz = zi - sz[jj];
            float d2 = dx * dx + dy * dy + dz * dz;
            float d  = sqrtf(fmaxf(d2, 1e-24f)) + 1e-8f;
            float cf = (d < 1.0f && jg != i) ? (1.0f - d) / (d * d) : 0.0f;
            ax = fmaf(cf, dx, ax);
            ay = fmaf(cf, dy, ay);
            az = fmaf(cf, dz, az);
        }
        __syncthreads();
    }
    atomicAdd(&clash_corr[i * 3 + 0], ax);
    atomicAdd(&clash_corr[i * 3 + 1], ay);
    atomicAdd(&clash_corr[i * 3 + 2], az);
}

// ---------------------------------------------------------------------------
// K5: out = v_x_raw - 0.1 * bond_w * bond_corr   (full overwrite of d_out)
// ---------------------------------------------------------------------------
__global__ void k_combine(const float* __restrict__ v_x_raw,
                          const float* __restrict__ bond_w,
                          const float* __restrict__ bond_corr,
                          float* __restrict__ out)
{
    int n = blockIdx.x * 256 + threadIdx.x;
    if (n >= NN) return;
    float bw = 0.1f * bond_w[n];
    out[n * 3 + 0] = v_x_raw[n * 3 + 0] - bw * bond_corr[n * 3 + 0];
    out[n * 3 + 1] = v_x_raw[n * 3 + 1] - bw * bond_corr[n * 3 + 1];
    out[n * 3 + 2] = v_x_raw[n * 3 + 2] - bw * bond_corr[n * 3 + 2];
}

// K6: subtract clash term at generated rows (gen_idx is unique -> no atomics)
__global__ void k_clash_apply(const int* __restrict__ gen_idx,
                              const float* __restrict__ clash_w,
                              const float* __restrict__ clash_corr,
                              float* __restrict__ out)
{
    int g = blockIdx.x * 256 + threadIdx.x;
    if (g >= GG) return;
    int idx = gen_idx[g];
    float cw = 0.05f * clash_w[idx];
    out[idx * 3 + 0] -= cw * clash_corr[g * 3 + 0];
    out[idx * 3 + 1] -= cw * clash_corr[g * 3 + 1];
    out[idx * 3 + 2] -= cw * clash_corr[g * 3 + 2];
}

// ---------------------------------------------------------------------------
extern "C" void kernel_launch(void* const* d_in, const int* in_sizes, int n_in,
                              void* d_out, int out_size, void* d_ws, size_t ws_size,
                              hipStream_t stream) {
    (void)in_sizes; (void)n_in; (void)out_size; (void)ws_size;

    const float*     v_x_raw = (const float*)d_in[0];
    const float*     H       = (const float*)d_in[1];
    const float*     x_t     = (const float*)d_in[2];
    const long long* edges   = (const long long*)d_in[3];
    // d_in[4] (gen_mask, bool) ignored: rebuilt from gen_idx to avoid dtype ambiguity
    const int*   gen_idx = (const int*)d_in[5];
    const float* W_proj  = (const float*)d_in[6];
    const float* b_proj  = (const float*)d_in[7];
    const float* w_bond  = (const float*)d_in[8];
    const float* b_bond  = (const float*)d_in[9];
    const float* w_val   = (const float*)d_in[10];
    const float* b_val   = (const float*)d_in[11];
    float* out = (float*)d_out;

    char*   ws         = (char*)d_ws;
    bf16_t* packB      = (bf16_t*)(ws + WS_PACKB);
    float*  bond_w     = (float*)(ws + WS_BONDW);
    float*  clash_w    = (float*)(ws + WS_CLASHW);
    int*    mask       = (int*)  (ws + WS_MASK);
    float*  bond_corr  = (float*)(ws + WS_BONDCORR);
    float*  xg         = (float*)(ws + WS_XG);
    float*  clash_corr = (float*)(ws + WS_CLASHCORR);

    k_zero<<<(NN * 3 + 255) / 256, 256, 0, stream>>>(bond_corr, clash_corr, mask);
    k_scatter<<<(GG + 255) / 256, 256, 0, stream>>>(gen_idx, x_t, mask, xg);
    k_pack_w<<<(16 * 16 * 32 + 255) / 256, 256, 0, stream>>>(W_proj, packB);

    k_gemm_sigmoid<<<NN / 128, 256, 131072, stream>>>(H, packB, b_proj,
                                                      w_bond, b_bond, w_val, b_val,
                                                      bond_w, clash_w);

    k_bond<<<(EE + 255) / 256, 256, 0, stream>>>(edges, x_t, mask, bond_corr);

    dim3 cgrid(GG / 256, SLICES);
    k_clash<<<cgrid, 256, 0, stream>>>(xg, clash_corr);

    k_combine<<<(NN + 255) / 256, 256, 0, stream>>>(v_x_raw, bond_w, bond_corr, out);
    k_clash_apply<<<(GG + 255) / 256, 256, 0, stream>>>(gen_idx, clash_w, clash_corr, out);
}
